// Net_SLSTM_15324443312129
// MI455X (gfx1250) — 8-bit matrix operand order repaired and checked by reading, compiled, not run on hardware
//
#include <hip/hip_runtime.h>
#include <cstdint>
#include <cstddef>

#define TT   800
#define BB   256
#define INW  14
#define HH   128
#define GG   512
#define NCG  8
#define NSD  7
#define DHH  64
#define BT   16
#define NWG  (BB/BT)
#define K1   192
#define K2   256
#define KT1  3
#define KT2  4
#define EPSV 1e-5f

#define OFF_W1    0
#define OFF_W2    98304
#define OFF_GATES 229376
#define OFF_SYN1  262144
#define OFF_MEM1  270336
#define OFF_SYN2  278528
#define OFF_MEM2  286720
#define OFF_B1    294912
#define OFF_B2    296960
#define OFF_A1    299008
#define OFF_A2    302080
#define OFF_XRAW  306176
#define LDS_BYTES 307968

typedef __attribute__((ext_vector_type(8))) int          v8i_t;
typedef __attribute__((ext_vector_type(8))) float        v8f_t;
typedef __attribute__((ext_vector_type(4))) unsigned int u32x4_t;
typedef __attribute__((ext_vector_type(8))) int          i32x8_t;
typedef __attribute__((ext_vector_type(4))) int          i32x4_t;

#if defined(__gfx1250__) && __has_builtin(__builtin_amdgcn_tensor_load_to_lds) && __has_builtin(__builtin_amdgcn_s_wait_tensorcnt)
#define USE_TDM 1
#else
#define USE_TDM 0
#endif

static __device__ __forceinline__ float sigf(float x) {
  return 1.0f / (1.0f + __expf(-x));
}
static __device__ __forceinline__ float tanh_fast(float x) {
  x = fminf(fmaxf(x, -30.0f), 30.0f);
  float e = __expf(-2.0f * x);
  return (1.0f - e) / (1.0f + e);
}

static __device__ __forceinline__ unsigned char f32_e4m3(float f) {
  union { float f; unsigned u; } v; v.f = f;
  unsigned s = (v.u >> 24) & 0x80u;
  float a = fabsf(f);
  if (!(a > 0.0f)) return (unsigned char)s;
  if (a > 448.0f) a = 448.0f;
  v.f = a;
  int e = (int)((v.u >> 23) & 0xffu) - 127;
  unsigned eb, m;
  if (e < -6) {
    int q = (int)(a * 512.0f + 0.5f);
    if (q > 7) { eb = 1u; m = 0u; }
    else       { eb = 0u; m = (unsigned)q; }
  } else {
    unsigned m23 = v.u & 0x7fffffu;
    m = m23 >> 20;
    if (m23 & 0x80000u) { ++m; if (m == 8u) { m = 0u; ++e; } }
    if (e > 8) { e = 8; m = 6u; }
    eb = (unsigned)(e + 7);
  }
  return (unsigned char)(s | (eb << 3) | m);
}

static __device__ __forceinline__ int a_off(int m, int k64) {
  int half = k64 >> 5;
  int r    = k64 & 31;
  int q    = r >> 2;
  int v    = half * 4 + ((q & 4) >> 1) + (q & 1);
  int lane = m + ((q & 2) << 3);
  return lane * 32 + v * 4 + (r & 3);
}

static __device__ __forceinline__ v8i_t load_frag(const char* base, int lane) {
  const int4* p = (const int4*)(base + lane * 32);
  int4 lo = p[0], hi = p[1];
  v8i_t r;
  r[0] = lo.x; r[1] = lo.y; r[2] = lo.z; r[3] = lo.w;
  r[4] = hi.x; r[5] = hi.y; r[6] = hi.z; r[7] = hi.w;
  return r;
}

#if USE_TDM
static __device__ __forceinline__ void tdm_issue_x(const float* xg, int t, int wg,
                                                   unsigned lds_addr) {
  unsigned long long ga =
      (unsigned long long)(uintptr_t)(xg + ((size_t)t * BB + (size_t)wg * BT) * INW);
  u32x4_t g0;
  g0[0] = 1u;
  g0[1] = lds_addr;
  g0[2] = (unsigned)ga;
  g0[3] = (unsigned)((ga >> 32) & 0x1FFFFFFull) | (2u << 30);
  i32x8_t g1;
  g1[0] = (2 << 16);
  g1[1] = (INW << 16);
  g1[2] = (BT  << 16);
  g1[3] = (INW << 16);
  g1[4] = BT;
  g1[5] = INW;
  g1[6] = 0;
  g1[7] = 0;
  i32x4_t z4; z4[0] = 0; z4[1] = 0; z4[2] = 0; z4[3] = 0;
  i32x8_t z8; z8[0] = 0; z8[1] = 0; z8[2] = 0; z8[3] = 0;
              z8[4] = 0; z8[5] = 0; z8[6] = 0; z8[7] = 0;
  __builtin_amdgcn_tensor_load_to_lds(g0, g1, z4, z4, z8, 0);
}
#endif

__global__ void __launch_bounds__(256, 1)
slstm_persistent_kernel(const float* __restrict__ x,
                        const float* __restrict__ Wih1, const float* __restrict__ Whh1,
                        const float* __restrict__ bih1, const float* __restrict__ bhh1,
                        const float* __restrict__ thr1,
                        const float* __restrict__ Wih2, const float* __restrict__ Whh2,
                        const float* __restrict__ bih2, const float* __restrict__ bhh2,
                        const float* __restrict__ thr2,
                        const float* __restrict__ bn_g, const float* __restrict__ bn_b,
                        const float* __restrict__ bn_m, const float* __restrict__ bn_v,
                        const float* __restrict__ Wfc,  const float* __restrict__ bfc,
                        const float* __restrict__ Wd1,  const float* __restrict__ bd1,
                        const float* __restrict__ thr_d,
                        const float* __restrict__ Wd2,  const float* __restrict__ bd2,
                        float* __restrict__ out) {
  extern __shared__ char sm[];
  const int tid  = threadIdx.x;
  const int wg   = blockIdx.x;
  const int lane = tid & 31;
  const int wv   = tid >> 5;
  const int ncol = lane & 15;
  const int khalf = (lane >> 4) << 4;

  unsigned char* w1raw = (unsigned char*)(sm + OFF_W1);
  unsigned char* w2raw = (unsigned char*)(sm + OFF_W2);
  float*         gates = (float*)(sm + OFF_GATES);
  float*         syn1  = (float*)(sm + OFF_SYN1);
  float*         mem1  = (float*)(sm + OFF_MEM1);
  float*         syn2  = (float*)(sm + OFF_SYN2);
  float*         mem2  = (float*)(sm + OFF_MEM2);
  float*         bias1 = (float*)(sm + OFF_B1);
  float*         bias2 = (float*)(sm + OFF_B2);
  unsigned char* a1s   = (unsigned char*)(sm + OFF_A1);
  unsigned char* a2s   = (unsigned char*)(sm + OFF_A2);

  for (int idx = tid; idx < K1 * GG; idx += 256) {
    int k = idx / GG, n = idx % GG;
    float w = 0.0f;
    if (k < INW)      w = Wih1[n * INW + k];
    else if (k >= 64) w = Whh1[n * HH + (k - 64)];
    w1raw[idx] = f32_e4m3(w);
  }
  for (int idx = tid; idx < K2 * GG; idx += 256) {
    int k = idx / GG, n = idx % GG;
    float w = (k < HH) ? Wih2[n * HH + k] : Whh2[n * HH + (k - HH)];
    w2raw[idx] = f32_e4m3(w);
  }
  for (int n = tid; n < GG; n += 256) {
    bias1[n] = bih1[n] + bhh1[n];
    bias2[n] = bih2[n] + bhh2[n];
  }
  for (int i = tid; i < BT * HH; i += 256) {
    syn1[i] = 0.0f; mem1[i] = 0.0f; syn2[i] = 0.0f; mem2[i] = 0.0f;
  }
  for (int i = tid; i < KT1 * 1024; i += 256) a1s[i] = 0;
  for (int i = tid; i < KT2 * 1024; i += 256) a2s[i] = 0;
  __syncthreads();

  v8i_t bf1[4][KT1];
  v8i_t bf2[4][KT2];
#pragma unroll
  for (int i = 0; i < 4; ++i) {
    int n = (wv * 4 + i) * 16 + ncol;
#pragma unroll
    for (int kt = 0; kt < KT1; ++kt) {
      v8i_t f;
#pragma unroll
      for (int d = 0; d < 8; ++d) {
        unsigned w = 0;
#pragma unroll
        for (int bb = 0; bb < 4; ++bb) {
          int e = d * 4 + bb;
          int k = kt * 64 + ((e >> 4) << 5) + ((e & 8) << 1) + (khalf >> 1) + (e & 7);
          w |= ((unsigned)w1raw[k * GG + n]) << (8 * bb);
        }
        f[d] = (int)w;
      }
      bf1[i][kt] = f;
    }
#pragma unroll
    for (int kt = 0; kt < KT2; ++kt) {
      v8i_t f;
#pragma unroll
      for (int d = 0; d < 8; ++d) {
        unsigned w = 0;
#pragma unroll
        for (int bb = 0; bb < 4; ++bb) {
          int e = d * 4 + bb;
          int k = kt * 64 + ((e >> 4) << 5) + ((e & 8) << 1) + (khalf >> 1) + (e & 7);
          w |= ((unsigned)w2raw[k * GG + n]) << (8 * bb);
        }
        f[d] = (int)w;
      }
      bf2[i][kt] = f;
    }
  }

  const int brow = tid >> 4;
  const int j0   = tid & 15;
  float bna[8], bnc[8], facc[8];
#pragma unroll
  for (int u = 0; u < 8; ++u) {
    int j = j0 + 16 * u;
    float sc = bn_g[j] * rsqrtf(bn_v[j] + EPSV);
    bna[u] = sc;
    bnc[u] = bn_b[j] - bn_m[j] * sc;
    facc[u] = 0.0f;
  }
  const float thr1v = thr1[0];
  const float thr2v = thr2[0];

#if USE_TDM
  const unsigned xlds = (unsigned)(unsigned long long)(uintptr_t)(sm + OFF_XRAW);
  if (wv == 0) {
    tdm_issue_x(x, 0, wg, xlds + 0);
    if (TT > 1) tdm_issue_x(x, 1, wg, xlds + 896);
    __builtin_amdgcn_s_wait_tensorcnt(1);
  }
  __syncthreads();
  if (tid < BT * INW) {
    int m = tid / INW, i = tid % INW;
    float xv = ((const float*)(sm + OFF_XRAW))[m * INW + i];
    a1s[a_off(m, i)] = f32_e4m3(xv);
  }
#else
  if (tid < BT * INW) {
    int m = tid / INW, i = tid % INW;
    float xv = x[((size_t)0 * BB + (size_t)wg * BT + m) * INW + i];
    a1s[a_off(m, i)] = f32_e4m3(xv);
  }
#endif
  __syncthreads();

  const int mhi = (lane >> 4) << 3;

  for (int t = 0; t < TT; ++t) {
#if USE_TDM
    if (wv == 0) {
      if (t + 2 < TT) {
        tdm_issue_x(x, t + 2, wg, xlds + (unsigned)(((t + 2) & 1) * 896));
        __builtin_amdgcn_s_wait_tensorcnt(1);
      } else {
        __builtin_amdgcn_s_wait_tensorcnt(0);
      }
    }
#endif
    {
      v8i_t af[KT1];
#pragma unroll
      for (int kt = 0; kt < KT1; ++kt)
        af[kt] = load_frag((const char*)a1s + kt * 1024, lane);
#pragma unroll
      for (int i = 0; i < 4; ++i) {
        v8f_t acc = {0.f, 0.f, 0.f, 0.f, 0.f, 0.f, 0.f, 0.f};
#pragma unroll
        for (int kt = 0; kt < KT1; ++kt)
          acc = __builtin_amdgcn_wmma_f32_16x16x64_fp8_fp8(
              af[kt], bf1[i][kt], (short)0, acc, false, false);
        int nb = (wv * 4 + i) * 16 + ncol;
#pragma unroll
        for (int r = 0; r < 8; ++r)
          gates[(mhi + r) * GG + nb] = acc[r];
      }
    }
    __syncthreads();

#pragma unroll
    for (int u = 0; u < 8; ++u) {
      int j = j0 + 16 * u;
      float gi = gates[brow * GG + j]          + bias1[j];
      float gf = gates[brow * GG + HH + j]     + bias1[HH + j];
      float gg = gates[brow * GG + 2 * HH + j] + bias1[2 * HH + j];
      float go = gates[brow * GG + 3 * HH + j] + bias1[3 * HH + j];
      float cs = syn1[brow * HH + j];
      float hm = mem1[brow * HH + j];
      float c  = sigf(gf) * cs + sigf(gi) * tanh_fast(gg);
      float h  = sigf(go) * tanh_fast(c) - ((hm > thr1v) ? thr1v : 0.0f);
      syn1[brow * HH + j] = c;
      mem1[brow * HH + j] = h;
      float spkbn = bnc[u] + (((h - thr1v) > 0.0f) ? bna[u] : 0.0f);
      a2s[(j >> 6) * 1024 + a_off(brow, j & 63)] = f32_e4m3(spkbn);
      int k1 = 64 + j;
      a1s[(k1 >> 6) * 1024 + a_off(brow, k1 & 63)] = f32_e4m3(h);
    }
    if (t + 1 < TT && tid < BT * INW) {
      int m = tid / INW, i = tid % INW;
#if USE_TDM
      float xv = ((const float*)(sm + OFF_XRAW + ((t + 1) & 1) * 896))[m * INW + i];
#else
      float xv = x[((size_t)(t + 1) * BB + (size_t)wg * BT + m) * INW + i];
#endif
      a1s[a_off(m, i)] = f32_e4m3(xv);
    }
    __syncthreads();

    {
      v8i_t af[KT2];
#pragma unroll
      for (int kt = 0; kt < KT2; ++kt)
        af[kt] = load_frag((const char*)a2s + kt * 1024, lane);
#pragma unroll
      for (int i = 0; i < 4; ++i) {
        v8f_t acc = {0.f, 0.f, 0.f, 0.f, 0.f, 0.f, 0.f, 0.f};
#pragma unroll
        for (int kt = 0; kt < KT2; ++kt)
          acc = __builtin_amdgcn_wmma_f32_16x16x64_fp8_fp8(
              af[kt], bf2[i][kt], (short)0, acc, false, false);
        int nb = (wv * 4 + i) * 16 + ncol;
#pragma unroll
        for (int r = 0; r < 8; ++r)
          gates[(mhi + r) * GG + nb] = acc[r];
      }
    }
    __syncthreads();

#pragma unroll
    for (int u = 0; u < 8; ++u) {
      int j = j0 + 16 * u;
      float gi = gates[brow * GG + j]          + bias2[j];
      float gf = gates[brow * GG + HH + j]     + bias2[HH + j];
      float gg = gates[brow * GG + 2 * HH + j] + bias2[2 * HH + j];
      float go = gates[brow * GG + 3 * HH + j] + bias2[3 * HH + j];
      float cs = syn2[brow * HH + j];
      float hm = mem2[brow * HH + j];
      float c  = sigf(gf) * cs + sigf(gi) * tanh_fast(gg);
      float h  = sigf(go) * tanh_fast(c) - ((hm > thr2v) ? thr2v : 0.0f);
      syn2[brow * HH + j] = c;
      mem2[brow * HH + j] = h;
      facc[u] += h;
      int k2 = 128 + j;
      a2s[(k2 >> 6) * 1024 + a_off(brow, k2 & 63)] = f32_e4m3(h);
    }
    __syncthreads();
  }

  float* feats = (float*)(sm + OFF_GATES);
  float* dh    = (float*)(sm + OFF_GATES + BT * HH * 4);
#pragma unroll
  for (int u = 0; u < 8; ++u)
    feats[brow * HH + j0 + 16 * u] = facc[u] * (1.0f / (float)TT);
  __syncthreads();

  if (tid < BT * NCG) {
    int b = tid / NCG, c = tid % NCG;
    float s = bfc[c];
    for (int k = 0; k < HH; ++k) s += feats[b * HH + k] * Wfc[c * HH + k];
    out[((size_t)wg * BT + b) * NCG + c] = s;
  }
  const float thrdv = thr_d[0];
  for (int idx = tid; idx < BT * DHH; idx += 256) {
    int b = idx / DHH, d = idx % DHH;
    float s = bd1[d];
    for (int k = 0; k < HH; ++k) s += feats[b * HH + k] * Wd1[d * HH + k];
    dh[idx] = ((s - thrdv) > 0.0f) ? 1.0f : 0.0f;
  }
  __syncthreads();
  if (tid < BT * NSD) {
    int b = tid / NSD, sdx = tid % NSD;
    float s = bd2[sdx];
    for (int k = 0; k < DHH; ++k) s += dh[b * DHH + k] * Wd2[sdx * DHH + k];
    out[(size_t)BB * NCG + ((size_t)wg * BT + b) * NSD + sdx] = s;
  }
}

extern "C" void kernel_launch(void* const* d_in, const int* in_sizes, int n_in,
                              void* d_out, int out_size, void* d_ws, size_t ws_size,
                              hipStream_t stream) {
  (void)in_sizes; (void)n_in; (void)out_size; (void)d_ws; (void)ws_size;
  const float* X    = (const float*)d_in[0];
  const float* Wih1 = (const float*)d_in[1];
  const float* Whh1 = (const float*)d_in[2];
  const float* bih1 = (const float*)d_in[3];
  const float* bhh1 = (const float*)d_in[4];
  const float* thr1 = (const float*)d_in[5];
  const float* Wih2 = (const float*)d_in[6];
  const float* Whh2 = (const float*)d_in[7];
  const float* bih2 = (const float*)d_in[8];
  const float* bhh2 = (const float*)d_in[9];
  const float* thr2 = (const float*)d_in[10];
  const float* bn_g = (const float*)d_in[11];
  const float* bn_b = (const float*)d_in[12];
  const float* bn_m = (const float*)d_in[13];
  const float* bn_v = (const float*)d_in[14];
  const float* Wfc  = (const float*)d_in[15];
  const float* bfc  = (const float*)d_in[16];
  const float* Wd1  = (const float*)d_in[17];
  const float* bd1  = (const float*)d_in[18];
  const float* thrd = (const float*)d_in[19];
  const float* Wd2  = (const float*)d_in[20];
  const float* bd2  = (const float*)d_in[21];
  float* out = (float*)d_out;

  (void)hipFuncSetAttribute((const void*)slstm_persistent_kernel,
                            hipFuncAttributeMaxDynamicSharedMemorySize, LDS_BYTES);
  slstm_persistent_kernel<<<NWG, 256, LDS_BYTES, stream>>>(
      X, Wih1, Whh1, bih1, bhh1, thr1, Wih2, Whh2, bih2, bhh2, thr2,
      bn_g, bn_b, bn_m, bn_v, Wfc, bfc, Wd1, bd1, thrd, Wd2, bd2, out);
}
